// AdaptiveExitAttention_74577812127935
// MI455X (gfx1250) — matrix operand order repaired and checked by reading, not compiled, not run on hardware
//
#include <hip/hip_runtime.h>

#define DEVI __device__ __forceinline__

typedef __attribute__((ext_vector_type(16))) __bf16 bf16x16;
typedef __attribute__((ext_vector_type(8)))  float  f32x8;
typedef __attribute__((ext_vector_type(4)))  int    v4i;

struct bfx8 { __bf16 v[8]; };

// Native fp32 -> bf16 conversion (lowers to v_cvt_pk_bf16_f32).
DEVI __bf16 f2bf(float f) { return (__bf16)f; }

// ---- async global->LDS copy (CDNA5 GLOBAL_LOAD_ASYNC_TO_LDS_B128) ----------
// Probe-discovered signature: param0 = v4i* in AS(1) (global), so pass
// address-space-qualified int4 pointers for src (AS1) and LDS dst (AS3).
typedef v4i __attribute__((address_space(1))) g_v4i;
typedef v4i __attribute__((address_space(3))) l_v4i;

#if __has_builtin(__builtin_amdgcn_global_load_async_to_lds_b128) && \
    __has_builtin(__builtin_amdgcn_s_wait_asynccnt)
#define USE_ASYNC_LDS 1
DEVI void async_copy16(const __bf16* g, __bf16* l) {
  __builtin_amdgcn_global_load_async_to_lds_b128(
      (g_v4i*)g, (l_v4i*)l, 0, 0);
}
DEVI void async_wait() { __builtin_amdgcn_s_wait_asynccnt(0); }
#else
DEVI void async_copy16(const __bf16* g, __bf16* l) {
  *reinterpret_cast<bfx8*>(l) = *reinterpret_cast<const bfx8*>(g);
}
DEVI void async_wait() {}
#endif

// A-fragment (16x32, MxK): lane = M, lanes 0-15 hold K=0..15,
// lanes 16-31 hold K=16..31.
DEVI bf16x16 load_a_frag(const __bf16* t, int ld, int r0, int k0, int lane) {
  int m  = lane & 15;
  int hh = lane >> 4;
  const __bf16* p = t + (size_t)(r0 + m) * ld + k0;
  bf16x16 a;
#pragma unroll
  for (int g = 0; g < 2; ++g)
#pragma unroll
    for (int i = 0; i < 8; ++i)
      a[g * 8 + i] = p[hh * 16 + g * 8 + i];
  return a;
}

// B-fragment (32x16, KxN) read from a transposed LDS tile Bt[n][k]:
// lane = N, lanes 0-15 hold K=0..15, lanes 16-31 hold K=16..31.
DEVI bf16x16 load_b_frag(const __bf16* t, int ld, int n0, int k0, int lane) {
  int n  = lane & 15;
  int hh = lane >> 4;
  const __bf16* p = t + (size_t)(n0 + n) * ld + k0 + hh * 16;
  bf16x16 b;
#pragma unroll
  for (int i = 0; i < 16; ++i) b[i] = p[i];
  return b;
}

DEVI f32x8 wmma_bf16(bf16x16 a, bf16x16 b, f32x8 c) {
  return __builtin_amdgcn_wmma_f32_16x16x32_bf16(false, a, false, b,
                                                 (short)0, c, false, false);
}

// ---------------------------------------------------------------------------
// QKV projection: Y = x @ W + b ; x fp32 [4096,1024], W fp32 [1024,1024].
// Output scattered as bf16 in [B=2, H=16, T=2048, Dk=64] layout.
// Block tile 128(M) x 64(N), BK=32, 256 threads = 8 waves, wave w owns rows
// [w*16, w*16+16) of the tile (4 WMMA accumulators across N).
// ---------------------------------------------------------------------------
__global__ __launch_bounds__(256)
void qkv_gemm(const float* __restrict__ A, const float* __restrict__ W,
              const float* __restrict__ bias, __bf16* __restrict__ dst) {
  constexpr int K = 1024, N = 1024, T = 2048, H = 16, DK = 64;
  constexpr int LDA = 40, LDB = 40;
  __shared__ __bf16 As[128 * LDA];
  __shared__ __bf16 Bt[64 * LDB];
  const int tid = threadIdx.x, wv = tid >> 5, lane = tid & 31;
  const int m0 = blockIdx.x * 128, n0 = blockIdx.y * 64;

  f32x8 acc[4] = {};

  for (int k0 = 0; k0 < K; k0 += 32) {
    __syncthreads();
    // A tile 128x32 fp32 -> bf16 LDS (1024 float4, 4 per thread)
#pragma unroll
    for (int i = 0; i < 4; ++i) {
      int idx = tid + i * 256;
      int r = idx >> 3, c4 = (idx & 7) * 4;
      const float4 f =
          *reinterpret_cast<const float4*>(A + (size_t)(m0 + r) * K + k0 + c4);
      __bf16* o = As + r * LDA + c4;
      o[0] = f2bf(f.x); o[1] = f2bf(f.y); o[2] = f2bf(f.z); o[3] = f2bf(f.w);
      if (k0 + 32 < K)
        __builtin_prefetch(A + (size_t)(m0 + r) * K + k0 + 32 + c4, 0, 1);
    }
    // W tile 32x64 fp32 -> transposed bf16 LDS Bt[n][k]
#pragma unroll
    for (int i = 0; i < 2; ++i) {
      int idx = tid + i * 256;
      int kk = idx >> 4, c4 = (idx & 15) * 4;
      const float4 f =
          *reinterpret_cast<const float4*>(W + (size_t)(k0 + kk) * N + n0 + c4);
      Bt[(c4 + 0) * LDB + kk] = f2bf(f.x);
      Bt[(c4 + 1) * LDB + kk] = f2bf(f.y);
      Bt[(c4 + 2) * LDB + kk] = f2bf(f.z);
      Bt[(c4 + 3) * LDB + kk] = f2bf(f.w);
    }
    __syncthreads();

    bf16x16 a = load_a_frag(As, LDA, wv * 16, 0, lane);
#pragma unroll
    for (int j = 0; j < 4; ++j) {
      bf16x16 b = load_b_frag(Bt, LDB, j * 16, 0, lane);
      acc[j] = wmma_bf16(a, b, acc[j]);
    }
  }

  // epilogue: bias + scatter to [B,H,T,Dk]
  const int hh = lane >> 4, nl = lane & 15;
#pragma unroll
  for (int j = 0; j < 4; ++j) {
    int n = n0 + j * 16 + nl;
    int h = n >> 6, d = n & 63;
    float bb = bias[n];
#pragma unroll
    for (int r = 0; r < 8; ++r) {
      int m = m0 + wv * 16 + hh * 8 + r;
      int bI = m >> 11, t = m & (T - 1);
      dst[(((size_t)bI * H + h) * T + t) * DK + d] = f2bf(acc[j][r] + bb);
    }
  }
}

// ---------------------------------------------------------------------------
// Flash attention. One block = 128 query rows of one (b,h). 8 waves, wave w
// owns q rows [w*16, w*16+16): softmax state never crosses waves, and the
// WMMA C-layout keeps distinct rows in distinct 16-lane halves, so row
// reductions are __shfl_xor with masks 1,2,4,8 only.
// Q,K,V bf16 [B,H,T,Dk]; output Y bf16 [B,T,C] (C = H*Dk) for the out-proj.
// Q/K tiles stream through GLOBAL_LOAD_ASYNC_TO_LDS (ASYNCcnt).
// ---------------------------------------------------------------------------
__global__ __launch_bounds__(256)
void flash_attn(const __bf16* __restrict__ Q, const __bf16* __restrict__ Kg,
                const __bf16* __restrict__ Vg, __bf16* __restrict__ Y) {
  constexpr int T = 2048, DK = 64, C = 1024;
  constexpr int LD = 72;                 // 144 B row pitch (16 B aligned)
  __shared__ __bf16 Qs[128 * LD];   // q rows (row-major, K = Dk)
  __shared__ __bf16 Ks[64 * LD];    // key rows: B-frag tile for S = Q K^T
  __shared__ __bf16 Vt[64 * LD];    // V transposed Vt[d][t]: B-frag tile for PV
  __shared__ __bf16 Ps[128 * LD];   // P round-trip (C-layout -> A-layout)

  const int tid = threadIdx.x, wv = tid >> 5, lane = tid & 31;
  const int hh = lane >> 4, nl = lane & 15;
  const int bh = blockIdx.x;             // b*16 + h
  const int bI = bh >> 4, h = bh & 15;
  const int qbase = blockIdx.y * 128;

  const __bf16* Qh = Q  + (size_t)bh * T * DK;
  const __bf16* Kh = Kg + (size_t)bh * T * DK;
  const __bf16* Vh = Vg + (size_t)bh * T * DK;

  // stage Q tile 128x64 (pure copy -> async to LDS)
#pragma unroll
  for (int i = 0; i < 4; ++i) {
    int idx = tid + i * 256;
    int r = idx >> 3, c8 = (idx & 7) * 8;
    async_copy16(Qh + (size_t)(qbase + r) * DK + c8, Qs + r * LD + c8);
  }

  f32x8 oacc[4] = {};
  float rmax[8], rsum[8];
#pragma unroll
  for (int r = 0; r < 8; ++r) { rmax[r] = -3.0e38f; rsum[r] = 0.f; }

  const int nkt = (qbase >> 6) + 2;   // causal bound: last k-tile with keys <= qmax
  for (int kb = 0; kb < nkt; ++kb) {
    const int kbase = kb * 64;
    __syncthreads();
    // stage K rows (async copy) + V transposed (manual scatter), 64x64 each
#pragma unroll
    for (int i = 0; i < 2; ++i) {
      int idx = tid + i * 256;
      int r = idx >> 3, c8 = (idx & 7) * 8;
      async_copy16(Kh + (size_t)(kbase + r) * DK + c8, Ks + r * LD + c8);
      bfx8 g = *reinterpret_cast<const bfx8*>(Vh + (size_t)(kbase + r) * DK + c8);
#pragma unroll
      for (int e = 0; e < 8; ++e) Vt[(c8 + e) * LD + r] = g.v[e];
    }
    async_wait();          // ASYNCcnt -> 0 (covers Q on first iteration too)
    __syncthreads();

    // S = Q K^T  (Dk=64 -> two k-steps of 32)
    f32x8 sacc[4] = {};
    bf16x16 a0 = load_a_frag(Qs, LD, wv * 16, 0, lane);
    bf16x16 a1 = load_a_frag(Qs, LD, wv * 16, 32, lane);
#pragma unroll
    for (int j = 0; j < 4; ++j) {
      bf16x16 b0 = load_b_frag(Ks, LD, j * 16, 0, lane);
      sacc[j] = wmma_bf16(a0, b0, sacc[j]);
      bf16x16 b1 = load_b_frag(Ks, LD, j * 16, 32, lane);
      sacc[j] = wmma_bf16(a1, b1, sacc[j]);
    }

    // online softmax on this wave's 16 rows (half-wave shuffles only)
#pragma unroll
    for (int r = 0; r < 8; ++r) {
      int m = qbase + wv * 16 + hh * 8 + r;
      float sv[4];
      float mloc = -3.0e38f;
#pragma unroll
      for (int j = 0; j < 4; ++j) {
        int n = kbase + j * 16 + nl;
        float v = (n <= m) ? sacc[j][r] * 0.125f : -3.0e38f;  // 1/sqrt(64)
        sv[j] = v;
        mloc = fmaxf(mloc, v);
      }
#pragma unroll
      for (int off = 1; off < 16; off <<= 1)
        mloc = fmaxf(mloc, __shfl_xor(mloc, off, 32));
      float nmax = fmaxf(rmax[r], mloc);
      float corr = __expf(rmax[r] - nmax);
      rmax[r] = nmax;
      float ps = 0.f;
#pragma unroll
      for (int j = 0; j < 4; ++j) {
        float p = __expf(sv[j] - nmax);
        ps += p;
        Ps[(wv * 16 + hh * 8 + r) * LD + j * 16 + nl] = f2bf(p);
      }
#pragma unroll
      for (int off = 1; off < 16; off <<= 1)
        ps += __shfl_xor(ps, off, 32);
      rsum[r] = rsum[r] * corr + ps;
#pragma unroll
      for (int j = 0; j < 4; ++j) oacc[j][r] *= corr;
    }

    // O += P @ V (wave-private Ps region: LDS ops are in-order within a wave)
    bf16x16 p0 = load_a_frag(Ps, LD, wv * 16, 0, lane);
    bf16x16 p1 = load_a_frag(Ps, LD, wv * 16, 32, lane);
#pragma unroll
    for (int j = 0; j < 4; ++j) {
      bf16x16 b0 = load_b_frag(Vt, LD, j * 16, 0, lane);
      oacc[j] = wmma_bf16(p0, b0, oacc[j]);
      bf16x16 b1 = load_b_frag(Vt, LD, j * 16, 32, lane);
      oacc[j] = wmma_bf16(p1, b1, oacc[j]);
    }
  }

  // normalize + store as [B,T,C] bf16
#pragma unroll
  for (int j = 0; j < 4; ++j) {
#pragma unroll
    for (int r = 0; r < 8; ++r) {
      int t = qbase + wv * 16 + hh * 8 + r;
      int c = h * DK + j * 16 + nl;
      Y[((size_t)bI * T + t) * C + c] = f2bf(oacc[j][r] / rsum[r]);
    }
  }
}

// ---------------------------------------------------------------------------
// Output projection: out = Y @ Wo + bo ; Y bf16 [4096,1024], out fp32.
// A tile staged with async-to-LDS (pure bf16 copy).
// ---------------------------------------------------------------------------
__global__ __launch_bounds__(256)
void out_gemm(const __bf16* __restrict__ A, const float* __restrict__ W,
              const float* __restrict__ bias, float* __restrict__ out) {
  constexpr int K = 1024, N = 1024;
  constexpr int LDA = 48, LDB = 40;      // LDA row pitch 96 B (16 B aligned)
  __shared__ __bf16 As[128 * LDA];
  __shared__ __bf16 Bt[64 * LDB];
  const int tid = threadIdx.x, wv = tid >> 5, lane = tid & 31;
  const int m0 = blockIdx.x * 128, n0 = blockIdx.y * 64;

  f32x8 acc[4] = {};

  for (int k0 = 0; k0 < K; k0 += 32) {
    __syncthreads();
    // A tile 128x32 bf16: async copy (512 x 16 B, 2 per thread)
#pragma unroll
    for (int i = 0; i < 2; ++i) {
      int idx = tid + i * 256;
      int r = idx >> 2, c8 = (idx & 3) * 8;
      async_copy16(A + (size_t)(m0 + r) * K + k0 + c8, As + r * LDA + c8);
      if (k0 + 32 < K)
        __builtin_prefetch(A + (size_t)(m0 + r) * K + k0 + 32 + c8, 0, 1);
    }
    // W tile 32x64 fp32 -> transposed bf16
#pragma unroll
    for (int i = 0; i < 2; ++i) {
      int idx = tid + i * 256;
      int kk = idx >> 4, c4 = (idx & 15) * 4;
      const float4 f =
          *reinterpret_cast<const float4*>(W + (size_t)(k0 + kk) * N + n0 + c4);
      Bt[(c4 + 0) * LDB + kk] = f2bf(f.x);
      Bt[(c4 + 1) * LDB + kk] = f2bf(f.y);
      Bt[(c4 + 2) * LDB + kk] = f2bf(f.z);
      Bt[(c4 + 3) * LDB + kk] = f2bf(f.w);
    }
    async_wait();
    __syncthreads();

    bf16x16 a = load_a_frag(As, LDA, wv * 16, 0, lane);
#pragma unroll
    for (int j = 0; j < 4; ++j) {
      bf16x16 b = load_b_frag(Bt, LDB, j * 16, 0, lane);
      acc[j] = wmma_bf16(a, b, acc[j]);
    }
  }

  const int hh = lane >> 4, nl = lane & 15;
#pragma unroll
  for (int j = 0; j < 4; ++j) {
    int n = n0 + j * 16 + nl;
    float bb = bias[n];
#pragma unroll
    for (int r = 0; r < 8; ++r) {
      int m = m0 + wv * 16 + hh * 8 + r;
      out[(size_t)m * N + n] = acc[j][r] + bb;
    }
  }
}

// ---------------------------------------------------------------------------
extern "C" void kernel_launch(void* const* d_in, const int* in_sizes, int n_in,
                              void* d_out, int out_size, void* d_ws, size_t ws_size,
                              hipStream_t stream) {
  const float* x  = (const float*)d_in[0];
  const float* Wq = (const float*)d_in[1];
  const float* bq = (const float*)d_in[2];
  const float* Wk = (const float*)d_in[3];
  const float* bk = (const float*)d_in[4];
  const float* Wv = (const float*)d_in[5];
  const float* bv = (const float*)d_in[6];
  const float* Wo = (const float*)d_in[7];
  const float* bo = (const float*)d_in[8];
  float* out = (float*)d_out;

  // workspace: 4 bf16 buffers of B*H*T*Dk = B*T*C = 4,194,304 elems (8 MB each)
  const size_t NEL = (size_t)2 * 2048 * 1024;
  __bf16* qb = (__bf16*)d_ws;
  __bf16* kb = qb + NEL;
  __bf16* vb = kb + NEL;
  __bf16* yb = vb + NEL;

  dim3 bb(256);
  dim3 gg(32, 16);   // M/128 = 32, N/64 = 16
  qkv_gemm<<<gg, bb, 0, stream>>>(x, Wq, bq, qb);
  qkv_gemm<<<gg, bb, 0, stream>>>(x, Wk, bk, kb);
  qkv_gemm<<<gg, bb, 0, stream>>>(x, Wv, bv, vb);
  flash_attn<<<dim3(32, 16), bb, 0, stream>>>(qb, kb, vb, yb);   // (B*H, T/128)
  out_gemm<<<gg, bb, 0, stream>>>(yb, Wo, bo, out);
}
